// CausalSelfAttention_73083163508885
// MI455X (gfx1250) — matrix operand order repaired and checked by reading, not compiled, not run on hardware
//
#include <hip/hip_runtime.h>

typedef __bf16 bf16_t;
typedef __attribute__((ext_vector_type(16))) __bf16 v16bf;
typedef __attribute__((ext_vector_type(8)))  __bf16 v8bf;
typedef __attribute__((ext_vector_type(8)))  float  v8f;
typedef unsigned short u16;

#define D_MODEL  2048
#define N_HEADS  16
#define HEAD_DIM 128
#define SEQ_T    2048
#define BATCH    2
#define MROWS    (BATCH * SEQ_T)   // 4096
#define QKV_N    (3 * D_MODEL)     // 6144
#define QSCALE   0.08838834764831845f   // 1/sqrt(128)

// ---------------- helpers ----------------

__device__ __forceinline__ u16 f2bf(float f) {
  unsigned int u = __float_as_uint(f);
  u += 0x7FFFu + ((u >> 16) & 1u);
  return (u16)(u >> 16);
}

// A-fragment (16x32 bf16, M in lanes, K split by lane-half):
// lane (m = lane&15, half = lane>>4) reads cols [k0+8h .. +7] and [k0+16+8h .. +7]
__device__ __forceinline__ v16bf load_a_frag(const u16* base, int ld, int row,
                                             int k0, int half) {
  const u16* p = base + (size_t)row * ld + k0 + half * 8;
  v8bf lo = *reinterpret_cast<const v8bf*>(p);
  v8bf hi = *reinterpret_cast<const v8bf*>(p + 16);
  v16bf a;
#pragma unroll
  for (int i = 0; i < 8; ++i) { a[i] = lo[i]; a[i + 8] = hi[i]; }
  return a;
}

// B-fragment (32x16 bf16, N in lanes): lane (n = lane&15, half) reads
// cols [k0+8h .. +7] and [k0+16+8h .. +7] of row n.
__device__ __forceinline__ v16bf load_b_frag(const u16* base, int ld, int row,
                                             int k0, int half) {
  const u16* p = base + (size_t)row * ld + k0 + half * 8;
  v8bf lo = *reinterpret_cast<const v8bf*>(p);
  v8bf hi = *reinterpret_cast<const v8bf*>(p + 16);
  v16bf b;
#pragma unroll
  for (int i = 0; i < 8; ++i) { b[i] = lo[i]; b[i + 8] = hi[i]; }
  return b;
}

__device__ __forceinline__ v8f wmma_bf16(v16bf a, v16bf b, v8f c) {
  return __builtin_amdgcn_wmma_f32_16x16x32_bf16(false, a, false, b,
                                                 (short)0, c, false, false);
}

// ---------------- kernel 1: fp32 -> bf16 convert ----------------

struct __align__(8) us4 { u16 x, y, z, w; };

__global__ void cvt_bf16_kernel(const float* __restrict__ src,
                                u16* __restrict__ dst, int nv) {
  for (int i = blockIdx.x * blockDim.x + threadIdx.x; i < nv;
       i += blockDim.x * gridDim.x) {
    float4 v = reinterpret_cast<const float4*>(src)[i];
    us4 o;
    o.x = f2bf(v.x); o.y = f2bf(v.y); o.z = f2bf(v.z); o.w = f2bf(v.w);
    reinterpret_cast<us4*>(dst)[i] = o;
  }
}

// ---------------- kernel 2: QKV GEMM + RoPE/split epilogue ----------------
// qkv[m,n] = sum_k x[m,k] * Wqkv[n,k];  M=4096, N=6144, K=2048.
// 64x32 wave tile (4x2 WMMA register blocking).
// Epilogue: split into Q (scaled), K, V^T with RoPE on heads 0..7 of Q,K.

__global__ __launch_bounds__(256) void qkv_gemm_rope_kernel(
    const u16* __restrict__ xbf, const u16* __restrict__ wbf,
    const float* __restrict__ cosT, const float* __restrict__ sinT,
    u16* __restrict__ qbf, u16* __restrict__ kbf, u16* __restrict__ vtbf) {
  const int wave = (blockIdx.x * blockDim.x + threadIdx.x) >> 5;
  const int lane = threadIdx.x & 31;
  const int half = lane >> 4, lm = lane & 15;
  const int ntiles = QKV_N / 32;                 // 192
  const int tm = wave / ntiles, tn = wave % ntiles;
  const int m0 = tm * 64, n0 = tn * 32;

  v8f acc[4][2] = {};
  for (int k0 = 0; k0 < D_MODEL; k0 += 32) {
    v16bf a[4], b[2];
#pragma unroll
    for (int i = 0; i < 4; ++i)
      a[i] = load_a_frag(xbf, D_MODEL, m0 + 16 * i + lm, k0, half);
#pragma unroll
    for (int j = 0; j < 2; ++j)
      b[j] = load_b_frag(wbf, D_MODEL, n0 + 16 * j + lm, k0, half);
#pragma unroll
    for (int i = 0; i < 4; ++i)
#pragma unroll
      for (int j = 0; j < 2; ++j)
        acc[i][j] = wmma_bf16(a[i], b[j], acc[i][j]);
  }

#pragma unroll
  for (int i = 0; i < 4; ++i)
#pragma unroll
    for (int j = 0; j < 2; ++j)
#pragma unroll
      for (int r = 0; r < 8; ++r) {
        const int m = m0 + i * 16 + r + 8 * half;   // global row (b*T + t)
        const int n = n0 + j * 16 + lm;             // global col in [0,6144)
        float val = acc[i][j][r];
        float partner = __shfl_xor(val, 1, 32);     // RoPE pair lives in lane^1
        const int b2 = m >> 11, t = m & 2047;
        const int sec = n >> 11;                    // 0=q 1=k 2=v
        const int hc = n & 2047;
        const int h = hc >> 7, d = hc & 127;
        float outv = val;
        if (sec < 2 && h < (N_HEADS / 2)) {
          float c = cosT[t * 64 + (d >> 1)];
          float s = sinT[t * 64 + (d >> 1)];
          outv = ((d & 1) == 0) ? (val * c - partner * s)
                                : (partner * s + val * c);
        }
        const size_t bh = (size_t)(b2 * N_HEADS + h);
        if (sec == 0) {
          qbf[(bh * SEQ_T + t) * HEAD_DIM + d] = f2bf(outv * QSCALE);
        } else if (sec == 1) {
          kbf[(bh * SEQ_T + t) * HEAD_DIM + d] = f2bf(outv);
        } else {
          vtbf[(bh * HEAD_DIM + d) * SEQ_T + t] = f2bf(outv);  // V transposed
        }
      }
}

// ---------------- kernel 3: flash attention ----------------
// One wave per (b,h, 16-row query tile). 32-key blocks, online softmax.

#define FA_WAVES 4

__global__ __launch_bounds__(FA_WAVES * 32) void flash_attn_kernel(
    const u16* __restrict__ qbf, const u16* __restrict__ kbf,
    const u16* __restrict__ vtbf, u16* __restrict__ attnbf) {
  __shared__ __align__(16) u16 pstage[FA_WAVES][16 * 32];

  const int wv = threadIdx.x >> 5;
  const int lane = threadIdx.x & 31;
  const int half = lane >> 4, lm = lane & 15;
  const int task = blockIdx.x * FA_WAVES + wv;     // 4096 tasks
  const int tq = task & 127;
  const int bh = task >> 7;                        // 32 (b,h) pairs
  const int t0 = tq * 16;

  const u16* Q  = qbf  + (size_t)bh * SEQ_T * HEAD_DIM;
  const u16* Kp = kbf  + (size_t)bh * SEQ_T * HEAD_DIM;
  const u16* Vt = vtbf + (size_t)bh * HEAD_DIM * SEQ_T;

  v16bf qf[4];
#pragma unroll
  for (int kk = 0; kk < 4; ++kk)
    qf[kk] = load_a_frag(Q, HEAD_DIM, t0 + lm, kk * 32, half);

  v8f o[8] = {};
  float mrow[8], lrow[8];
#pragma unroll
  for (int r = 0; r < 8; ++r) { mrow[r] = -1e30f; lrow[r] = 0.0f; }

  u16* ps = pstage[wv];

  for (int s0 = 0; s0 <= t0 + 15; s0 += 32) {
    // S = Q K^T  (two 16x16 tiles covering 32 keys)
    v8f S0 = {}, S1 = {};
#pragma unroll
    for (int kk = 0; kk < 4; ++kk) {
      v16bf b0 = load_b_frag(Kp, HEAD_DIM, s0 + lm,      kk * 32, half);
      v16bf b1 = load_b_frag(Kp, HEAD_DIM, s0 + 16 + lm, kk * 32, half);
      S0 = wmma_bf16(qf[kk], b0, S0);
      S1 = wmma_bf16(qf[kk], b1, S1);
    }

    float alpha[8];
#pragma unroll
    for (int r = 0; r < 8; ++r) {
      const int t = t0 + r + 8 * half;
      S0[r] = (s0 + lm      > t) ? -3.0e38f : S0[r];   // causal mask
      S1[r] = (s0 + 16 + lm > t) ? -3.0e38f : S1[r];
      float v = fmaxf(S0[r], S1[r]);
      v = fmaxf(v, __shfl_xor(v, 1, 32));
      v = fmaxf(v, __shfl_xor(v, 2, 32));
      v = fmaxf(v, __shfl_xor(v, 4, 32));
      v = fmaxf(v, __shfl_xor(v, 8, 32));              // row max within half
      const float mnew = fmaxf(mrow[r], v);
      alpha[r] = __expf(mrow[r] - mnew);
      mrow[r] = mnew;
      float p0 = __expf(S0[r] - mnew);
      float p1 = __expf(S1[r] - mnew);
      S0[r] = p0; S1[r] = p1;
      float rs = p0 + p1;
      rs += __shfl_xor(rs, 1, 32);
      rs += __shfl_xor(rs, 2, 32);
      rs += __shfl_xor(rs, 4, 32);
      rs += __shfl_xor(rs, 8, 32);
      lrow[r] = lrow[r] * alpha[r] + rs;
    }

#pragma unroll
    for (int dch = 0; dch < 8; ++dch)
#pragma unroll
      for (int r = 0; r < 8; ++r) o[dch][r] *= alpha[r];

    // stage P (C-layout) to LDS as bf16, reload as A-fragment layout
#pragma unroll
    for (int r = 0; r < 8; ++r) {
      const int row = r + 8 * half;
      ps[row * 32 + lm]      = f2bf(S0[r]);
      ps[row * 32 + 16 + lm] = f2bf(S1[r]);
    }
    __asm__ volatile("s_wait_dscnt 0" ::: "memory");
    v16bf pf;
    {
      const u16* pp = ps + lm * 32 + half * 8;
      v8bf lo = *reinterpret_cast<const v8bf*>(pp);
      v8bf hi = *reinterpret_cast<const v8bf*>(pp + 16);
#pragma unroll
      for (int i = 0; i < 8; ++i) { pf[i] = lo[i]; pf[i + 8] = hi[i]; }
    }

    // O += P V  (V stored transposed: contiguous B-fragments)
#pragma unroll
    for (int dch = 0; dch < 8; ++dch) {
      v16bf bv = load_b_frag(Vt, SEQ_T, dch * 16 + lm, s0, half);
      o[dch] = wmma_bf16(pf, bv, o[dch]);
    }
  }

  // normalize + write attn output in (B, T, D) layout as bf16
  const int b = bh >> 4, h = bh & 15;
#pragma unroll
  for (int dch = 0; dch < 8; ++dch)
#pragma unroll
    for (int r = 0; r < 8; ++r) {
      const int t = t0 + r + 8 * half;
      const int d = dch * 16 + lm;
      float val = o[dch][r] / lrow[r];
      attnbf[((size_t)(b * SEQ_T) + t) * D_MODEL + h * HEAD_DIM + d] = f2bf(val);
    }
}

// ---------------- kernel 4: output GEMM (fp32 result) ----------------
// out[m,n] = sum_k attn[m,k] * Wout[n,k];  M=4096, N=2048, K=2048.
// 64x32 wave tile (4x2 WMMA register blocking).

__global__ __launch_bounds__(256) void out_gemm_kernel(
    const u16* __restrict__ abf, const u16* __restrict__ wbf,
    float* __restrict__ out) {
  const int wave = (blockIdx.x * blockDim.x + threadIdx.x) >> 5;
  const int lane = threadIdx.x & 31;
  const int half = lane >> 4, lm = lane & 15;
  const int ntiles = D_MODEL / 32;               // 64
  const int tm = wave / ntiles, tn = wave % ntiles;
  const int m0 = tm * 64, n0 = tn * 32;

  v8f acc[4][2] = {};
  for (int k0 = 0; k0 < D_MODEL; k0 += 32) {
    v16bf a[4], b[2];
#pragma unroll
    for (int i = 0; i < 4; ++i)
      a[i] = load_a_frag(abf, D_MODEL, m0 + 16 * i + lm, k0, half);
#pragma unroll
    for (int j = 0; j < 2; ++j)
      b[j] = load_b_frag(wbf, D_MODEL, n0 + 16 * j + lm, k0, half);
#pragma unroll
    for (int i = 0; i < 4; ++i)
#pragma unroll
      for (int j = 0; j < 2; ++j)
        acc[i][j] = wmma_bf16(a[i], b[j], acc[i][j]);
  }

#pragma unroll
  for (int i = 0; i < 4; ++i)
#pragma unroll
    for (int j = 0; j < 2; ++j)
#pragma unroll
      for (int r = 0; r < 8; ++r)
        out[(size_t)(m0 + i * 16 + r + 8 * half) * D_MODEL + n0 + j * 16 + lm] =
            acc[i][j][r];
}

// ---------------- host launcher ----------------

extern "C" void kernel_launch(void* const* d_in, const int* in_sizes, int n_in,
                              void* d_out, int out_size, void* d_ws,
                              size_t ws_size, hipStream_t stream) {
  (void)in_sizes; (void)n_in; (void)out_size; (void)ws_size;
  const float* x    = (const float*)d_in[0];
  const float* Wqkv = (const float*)d_in[1];
  const float* Wout = (const float*)d_in[2];
  const float* cosT = (const float*)d_in[3];
  const float* sinT = (const float*)d_in[4];
  // d_in[5] rope_mask == (arange(16) < 8), hardcoded in the kernel.
  float* out = (float*)d_out;

  char* w = (char*)d_ws;
  const size_t szX    = (size_t)MROWS * D_MODEL;           // 8.4M elems
  const size_t szWqkv = (size_t)QKV_N * D_MODEL;           // 12.6M
  const size_t szWout = (size_t)D_MODEL * D_MODEL;         // 4.2M
  const size_t szHead = (size_t)BATCH * N_HEADS * SEQ_T * HEAD_DIM;  // 8.4M

  u16* xbf    = (u16*)w;  w += szX    * 2;
  u16* wqkvbf = (u16*)w;  w += szWqkv * 2;
  u16* woutbf = (u16*)w;  w += szWout * 2;
  u16* qbf    = (u16*)w;  w += szHead * 2;
  u16* kbf    = (u16*)w;  w += szHead * 2;
  u16* vtbf   = (u16*)w;  w += szHead * 2;
  u16* attnbf = (u16*)w;  w += szX    * 2;

  // 1) converts
  {
    int nv = (int)(szX / 4);
    cvt_bf16_kernel<<<(nv + 255) / 256, 256, 0, stream>>>(x, xbf, nv);
    nv = (int)(szWqkv / 4);
    cvt_bf16_kernel<<<(nv + 255) / 256, 256, 0, stream>>>(Wqkv, wqkvbf, nv);
    nv = (int)(szWout / 4);
    cvt_bf16_kernel<<<(nv + 255) / 256, 256, 0, stream>>>(Wout, woutbf, nv);
  }

  // 2) QKV GEMM + RoPE/split: (4096/64)*(6144/32)=12288 waves, 8 waves/block
  qkv_gemm_rope_kernel<<<12288 / 8, 256, 0, stream>>>(
      xbf, wqkvbf, cosT, sinT, qbf, kbf, vtbf);

  // 3) flash attention: 4096 wave-tasks, 4 waves/block
  flash_attn_kernel<<<4096 / FA_WAVES, FA_WAVES * 32, 0, stream>>>(
      qbf, kbf, vtbf, attnbf);

  // 4) output GEMM: (4096/64)*(2048/32)=4096 waves, 8 waves/block
  out_gemm_kernel<<<4096 / 8, 256, 0, stream>>>(attnbf, woutbf, out);
}
